// LSTM_26929444946222
// MI455X (gfx1250) — matrix operand order repaired and checked by reading, not compiled, not run on hardware
//
#include <hip/hip_runtime.h>
#include <hip/hip_bf16.h>
#include <math.h>

typedef __bf16 bf16_t;
typedef __attribute__((ext_vector_type(16))) __bf16 v16bf;
typedef __attribute__((ext_vector_type(8)))  __bf16 v8bf;
typedef __attribute__((ext_vector_type(8)))  float  v8f;

#define DEVFN static __device__ __forceinline__

// ---------------- problem constants ----------------
constexpr int kB = 64, kT = 512, kD = 256, kH = 512, kO = 64;
constexpr int kFourH = 4 * kH;        // 2048 gate columns
constexpr int kNB4   = kFourH / 16;   // 128 packed column tiles
constexpr int kK1 = kD + kH;          // 768
constexpr int kK2 = 2 * kH;           // 1024
constexpr int kKB1 = kK1 / 32;        // 24 k-blocks (layer1)
constexpr int kKB2 = kK2 / 32;        // 32 k-blocks (layer2)
constexpr unsigned kNWG = 32;         // persistent workgroups (one per nh tile)
constexpr int kTPB = 128;             // 4 waves -> 4 batch tiles
// LDS: per-nh weight slice, chunk-major blocks of 512 bf16 (1 KB)
constexpr int kL1Blocks = kKB1 * 4;         // 96 blocks
constexpr int kL2Blocks = kKB2 * 4;         // 128 blocks
constexpr size_t kLdsBytes = (size_t)(kL1Blocks + kL2Blocks) * 512 * 2; // 229376

// ---------------- WMMA helpers ----------------
DEVFN v8f wmma_bf16(v16bf a, v16bf b, v8f c) {
  // (neg_a, A, neg_b, B, c_mod, C, reuse_a, reuse_b)
  return __builtin_amdgcn_wmma_f32_16x16x32_bf16(false, a, false, b, (short)0, c,
                                                 false, false);
}

DEVFN float sigmoid_(float x) { return 1.0f / (1.0f + __expf(-x)); }
DEVFN float tanh_(float x)    { return 1.0f - 2.0f / (__expf(2.0f * x) + 1.0f); }

// A-fragment (16x32 MxK, bf16): lane<16 holds row M=lane, K {k..k+15};
// lane>=16 holds row M=lane-16, K {k+16..k+31}.  Source is plain
// row-major [rows][stride] bf16, so two contiguous 16B chunks per lane.
DEVFN v16bf load_a_tile(const bf16_t* __restrict__ base, int stride, int m0,
                        int lane, int kbase) {
  int row = m0 + (lane & 15);
  int c0  = kbase + ((lane >> 4) << 4);
  const bf16_t* p = base + (size_t)row * stride + c0;
  v8bf lo = *(const v8bf*)(p);
  v8bf hi = *(const v8bf*)(p + 8);
  return __builtin_shufflevector(lo, hi, 0, 1, 2, 3, 4, 5, 6, 7,
                                 8, 9, 10, 11, 12, 13, 14, 15);
}

// B-fragment from LDS, chunk-major: block = 2 planes of 32 lanes x 8 bf16.
// Lane reads 16B at lane*16 in each plane -> conflict-free ds_load_b128 x2.
DEVFN v16bf load_b_lds(const bf16_t* lbase, int blk, int lane) {
  const bf16_t* p = lbase + (size_t)blk * 512 + lane * 8;
  v8bf lo = *(const v8bf*)(p);
  v8bf hi = *(const v8bf*)(p + 256);
  return __builtin_shufflevector(lo, hi, 0, 1, 2, 3, 4, 5, 6, 7,
                                 8, 9, 10, 11, 12, 13, 14, 15);
}

// ---------------- device-wide barrier (persistent grid) ----------------
DEVFN void grid_sync(unsigned* cnt, unsigned* gen, unsigned nwg) {
  __syncthreads();
  if (threadIdx.x == 0) {
    __threadfence();
    unsigned g = __hip_atomic_load(gen, __ATOMIC_ACQUIRE, __HIP_MEMORY_SCOPE_AGENT);
    unsigned a = __hip_atomic_fetch_add(cnt, 1u, __ATOMIC_ACQ_REL,
                                        __HIP_MEMORY_SCOPE_AGENT);
    if (a == nwg - 1u) {
      __hip_atomic_store(cnt, 0u, __ATOMIC_RELAXED, __HIP_MEMORY_SCOPE_AGENT);
      __hip_atomic_fetch_add(gen, 1u, __ATOMIC_RELEASE, __HIP_MEMORY_SCOPE_AGENT);
    } else {
      while (__hip_atomic_load(gen, __ATOMIC_ACQUIRE, __HIP_MEMORY_SCOPE_AGENT) == g)
        __builtin_amdgcn_s_sleep(8);
    }
    __threadfence();
  }
  __syncthreads();
}

// ---------------- one-time prep kernels ----------------
// Pack fp32 W[K][2048] (row-major) into bf16 B-fragment blocks, CHUNK-MAJOR:
//   elem idx = (kb*128 + nb4)*512 + chunk*256 + lane*8 + e   (s = chunk*8 + e)
//   lane<16 : K = kb*32 + s,      N = nb4*16 + lane
//   lane>=16: K = kb*32 + 16 + s, N = nb4*16 + (lane-16)
__global__ void pack_w_kernel(const float* __restrict__ W, bf16_t* __restrict__ out,
                              int kblocks) {
  size_t idx = (size_t)blockIdx.x * blockDim.x + threadIdx.x;
  size_t total = (size_t)kblocks * kNB4 * 512;
  if (idx >= total) return;
  int e     = (int)(idx & 7);
  int lane  = (int)((idx >> 3) & 31);
  int chunk = (int)((idx >> 8) & 1);
  size_t blk = idx >> 9;
  int nb4 = (int)(blk & (kNB4 - 1));
  int kb  = (int)(blk >> 7);
  int s = chunk * 8 + e;
  int n = nb4 * 16 + (lane & 15);
  int klocal = (lane < 16) ? s : (16 + s);
  int k = kb * 32 + klocal;
  out[idx] = (bf16_t)W[(size_t)k * kFourH + n];
}

__global__ void cvt_bf16_kernel(const float* __restrict__ in,
                                bf16_t* __restrict__ out, size_t n) {
  size_t i = (size_t)blockIdx.x * blockDim.x + threadIdx.x;
  if (i < n) out[i] = (bf16_t)in[i];
}

// ---------------- fused LSTM cell update ----------------
DEVFN void cell_update(v8f ai, v8f aj, v8f af, v8f ao,
                       const float* __restrict__ bias, float* __restrict__ c,
                       bf16_t* __restrict__ hw, float* __restrict__ hf,
                       int rowbase, int col) {
  float bi = bias[col];
  float bj = bias[kH + col];
  float bf = bias[2 * kH + col];
  float bo = bias[3 * kH + col];
#pragma unroll
  for (int r = 0; r < 8; ++r) {
    size_t idx = (size_t)(rowbase + r) * kH + col;
    float cold = c[idx];
    float iv = sigmoid_(ai[r] + bi);
    float jv = tanh_(aj[r] + bj);
    float fv = sigmoid_(af[r] + bf + 1.0f);   // FORGET_BIAS
    float ov = sigmoid_(ao[r] + bo);
    float cn = cold * fv + iv * jv;
    float hn = tanh_(cn) * ov;
    c[idx] = cn;
    hw[idx] = (bf16_t)hn;
    if (hf) hf[idx] = hn;
  }
}

// 4 gate WMMAs sharing one A fragment; B from LDS (blk = KB*4 + gate)
#define GATE_STEP(LW, KB)                                      \
  ai = wmma_bf16(a, load_b_lds(LW, (KB) * 4 + 0, lane), ai);   \
  aj = wmma_bf16(a, load_b_lds(LW, (KB) * 4 + 1, lane), aj);   \
  af = wmma_bf16(a, load_b_lds(LW, (KB) * 4 + 2, lane), af);   \
  ao = wmma_bf16(a, load_b_lds(LW, (KB) * 4 + 3, lane), ao);

#define ZERO8 {0.f, 0.f, 0.f, 0.f, 0.f, 0.f, 0.f, 0.f}

// ---------------- persistent 2-layer LSTM ----------------
__global__ __launch_bounds__(kTPB, 1)
void lstm_persistent_kernel(const bf16_t* __restrict__ xb,   // [B][T][D] bf16
                            const bf16_t* __restrict__ w1p,  // packed chunk-major
                            const bf16_t* __restrict__ w2p,  // packed chunk-major
                            const float* __restrict__ b1,
                            const float* __restrict__ b2,
                            bf16_t* __restrict__ h1,         // [2][B*H]
                            bf16_t* __restrict__ h2,         // [2][B*H]
                            float* __restrict__ c1,          // [B*H]
                            float* __restrict__ c2,          // [B*H]
                            float* __restrict__ h2f,         // [B*H] fp32 copy
                            const float* __restrict__ Wout,  // [H][O]
                            const float* __restrict__ bout,  // [O]
                            float* __restrict__ out,         // [B][O]
                            unsigned* cnt, unsigned* gen) {
  extern __shared__ char smem_raw[];
  bf16_t* lw1 = (bf16_t*)smem_raw;                          // 96 KB
  bf16_t* lw2 = lw1 + (size_t)kL1Blocks * 512;              // 128 KB

  const int lane = threadIdx.x & 31;
  const int nh = blockIdx.x;                 // hidden tile 0..31 (per block)
  const int mt = threadIdx.x >> 5;           // batch tile  0..3  (per wave)
  const int col = nh * 16 + (lane & 15);
  const int rowbase = mt * 16 + ((lane >> 4) & 1) * 8;  // C/D frag row base

  // ---- stage this block's weight slice (all gates, all kb) into LDS ----
  {
    const uint4* s1 = (const uint4*)w1p;     // 64 uint4 per 1KB block
    const uint4* s2 = (const uint4*)w2p;
    uint4* d1 = (uint4*)lw1;
    uint4* d2 = (uint4*)lw2;
    for (int i = threadIdx.x; i < kL1Blocks * 64; i += kTPB) {
      int blk = i >> 6, w = i & 63;          // blk = kb*4 + g
      int kb = blk >> 2, g = blk & 3;
      d1[i] = s1[((size_t)kb * kNB4 + g * 32 + nh) * 64 + w];
    }
    for (int i = threadIdx.x; i < kL2Blocks * 64; i += kTPB) {
      int blk = i >> 6, w = i & 63;
      int kb = blk >> 2, g = blk & 3;
      d2[i] = s2[((size_t)kb * kNB4 + g * 32 + nh) * 64 + w];
    }
    __syncthreads();
  }

  int cur = 0;
  for (int t = 0; t < kT; ++t) {
    // ---------- layer 1: z = [x_t, h1] @ W1 ----------
    {
      v8f ai = ZERO8, aj = ZERO8, af = ZERO8, ao = ZERO8;
      const bf16_t* xrow = xb + (size_t)t * kD;          // batch stride T*D
#pragma unroll
      for (int kb = 0; kb < kD / 32; ++kb) {             // K in x
        v16bf a = load_a_tile(xrow, kT * kD, mt * 16, lane, kb * 32);
        GATE_STEP(lw1, kb)
      }
      const bf16_t* h1rd = h1 + (size_t)cur * (kB * kH);
#pragma unroll 4
      for (int kb = 0; kb < kH / 32; ++kb) {             // K in h1
        v16bf a = load_a_tile(h1rd, kH, mt * 16, lane, kb * 32);
        GATE_STEP(lw1, kD / 32 + kb)
      }
      bf16_t* h1wr = h1 + (size_t)(cur ^ 1) * (kB * kH);
      cell_update(ai, aj, af, ao, b1, c1, h1wr, nullptr, rowbase, col);
    }
    grid_sync(cnt, gen, kNWG);
    // ---------- layer 2: z = [h1_t, h2] @ W2 ----------
    {
      v8f ai = ZERO8, aj = ZERO8, af = ZERO8, ao = ZERO8;
      const bf16_t* h1rd = h1 + (size_t)(cur ^ 1) * (kB * kH);
#pragma unroll 4
      for (int kb = 0; kb < kH / 32; ++kb) {             // K in h1_t
        v16bf a = load_a_tile(h1rd, kH, mt * 16, lane, kb * 32);
        GATE_STEP(lw2, kb)
      }
      const bf16_t* h2rd = h2 + (size_t)cur * (kB * kH);
#pragma unroll 4
      for (int kb = 0; kb < kH / 32; ++kb) {             // K in h2_{t-1}
        v16bf a = load_a_tile(h2rd, kH, mt * 16, lane, kb * 32);
        GATE_STEP(lw2, kH / 32 + kb)
      }
      bf16_t* h2wr = h2 + (size_t)(cur ^ 1) * (kB * kH);
      cell_update(ai, aj, af, ao, b2, c2, h2wr, h2f, rowbase, col);
    }
    cur ^= 1;
  }
  grid_sync(cnt, gen, kNWG);
  // ---------- final projection: out = h2_T @ W_out + b_out ----------
  int tid = blockIdx.x * kTPB + threadIdx.x;  // exactly 4096 = B*O threads
  int b = tid >> 6, o = tid & 63;
  float s = bout[o];
  for (int k = 0; k < kH; ++k)
    s = fmaf(h2f[(size_t)b * kH + k], Wout[(size_t)k * kO + o], s);
  out[(size_t)b * kO + o] = s;
}

// ---------------- host launcher ----------------
extern "C" void kernel_launch(void* const* d_in, const int* in_sizes, int n_in,
                              void* d_out, int out_size, void* d_ws, size_t ws_size,
                              hipStream_t stream) {
  (void)in_sizes; (void)n_in; (void)out_size; (void)ws_size;
  const float* x    = (const float*)d_in[0];
  const float* W1   = (const float*)d_in[1];
  const float* b1   = (const float*)d_in[2];
  const float* W2   = (const float*)d_in[3];
  const float* b2   = (const float*)d_in[4];
  const float* Wout = (const float*)d_in[5];
  const float* bout = (const float*)d_in[6];
  float* out = (float*)d_out;

  char* ws = (char*)d_ws;
  size_t off = 0;
  bf16_t* w1p = (bf16_t*)(ws + off); off += (size_t)kK1 * kFourH * 2;  // 3.0 MB
  bf16_t* w2p = (bf16_t*)(ws + off); off += (size_t)kK2 * kFourH * 2;  // 4.0 MB
  bf16_t* xb  = (bf16_t*)(ws + off); off += (size_t)kB * kT * kD * 2;  // 16 MB
  char* state = ws + off;
  size_t sBH16 = (size_t)kB * kH * 2;   // bf16 [B*H]
  size_t sBH32 = (size_t)kB * kH * 4;   // fp32 [B*H]
  bf16_t* h1  = (bf16_t*)(state);                       // [2][B*H]
  bf16_t* h2  = (bf16_t*)(state + 2 * sBH16);           // [2][B*H]
  float*  c1  = (float*)(state + 4 * sBH16);
  float*  c2  = (float*)(state + 4 * sBH16 + sBH32);
  float*  h2f = (float*)(state + 4 * sBH16 + 2 * sBH32);
  unsigned* sync = (unsigned*)(state + 4 * sBH16 + 3 * sBH32);
  size_t stateBytes = 4 * sBH16 + 3 * sBH32 + 256;

  // zero recurrent state + barrier (capture-safe)
  hipMemsetAsync(state, 0, stateBytes, stream);

  // pack weights to chunk-major bf16 B-fragments; convert x to bf16
  {
    size_t t1 = (size_t)kKB1 * kNB4 * 512;
    size_t t2 = (size_t)kKB2 * kNB4 * 512;
    size_t tx = (size_t)kB * kT * kD;
    pack_w_kernel<<<dim3((unsigned)((t1 + 255) / 256)), 256, 0, stream>>>(W1, w1p, kKB1);
    pack_w_kernel<<<dim3((unsigned)((t2 + 255) / 256)), 256, 0, stream>>>(W2, w2p, kKB2);
    cvt_bf16_kernel<<<dim3((unsigned)((tx + 255) / 256)), 256, 0, stream>>>(x, xb, tx);
  }

  // persistent fused 2-layer LSTM + projection (one block per hidden tile,
  // 224 KB dynamic LDS holds that tile's full weight slice for both layers)
  lstm_persistent_kernel<<<dim3(kNWG), dim3(kTPB), kLdsBytes, stream>>>(
      xb, w1p, w2p, b1, b2, h1, h2, c1, c2, h2f, Wout, bout, out,
      sync, sync + 1);
}
